// CrossAttnRoPE_12601434046895
// MI455X (gfx1250) — matrix operand order repaired and checked by reading, not compiled, not run on hardware
//
#include <hip/hip_runtime.h>
#include <hip/hip_bf16.h>

typedef __bf16 v16bf __attribute__((ext_vector_type(16)));
typedef __bf16 v8bf  __attribute__((ext_vector_type(8)));
typedef float  v8f   __attribute__((ext_vector_type(8)));

#define WMMA_BF16(A, B, C) \
  __builtin_amdgcn_wmma_f32_16x16x32_bf16(false, (A), false, (B), (short)0, (C), false, false)

static __device__ inline v16bf ldA16(const __bf16* p0, const __bf16* p1) {
  union { v16bf v; v8bf h[2]; } u;
  u.h[0] = *(const v8bf*)p0;
  u.h[1] = *(const v8bf*)p1;
  return u.v;
}

// ---------------------------------------------------------------- f32 -> bf16
__global__ void cvt_bf16_kernel(const float* __restrict__ in, __bf16* __restrict__ out, int n) {
  int i = blockIdx.x * blockDim.x + threadIdx.x;
  if (i < n) out[i] = (__bf16)in[i];
}

// ------------------------------------------------- C[M,N] f32 = A[M,K] * W[N,K]^T
// 8 waves / block; each wave computes a 16x64 tile (4 N-tiles sharing one
// A-fragment per k-chunk -> 4 back-to-back WMMAs). Block tile = 128x64.
__global__ __launch_bounds__(256) void gemm_bf16_f32_kernel(
    const __bf16* __restrict__ A, const __bf16* __restrict__ W,
    float* __restrict__ C, int M, int N, int K) {
  const int lane  = threadIdx.x & 31;
  const int wave  = threadIdx.x >> 5;
  const int nl    = lane & 15;
  const int khalf = lane >> 4;
  const int n0 = blockIdx.x * 64;
  const int m0 = blockIdx.y * 128 + wave * 16;

  const __bf16* arow = A + (size_t)(m0 + nl) * K;          // A frag: lane = M row
  const __bf16* w0 = W + (size_t)(n0 +  0 + nl) * K;       // B frags: lane = N col
  const __bf16* w1 = W + (size_t)(n0 + 16 + nl) * K;
  const __bf16* w2 = W + (size_t)(n0 + 32 + nl) * K;
  const __bf16* w3 = W + (size_t)(n0 + 48 + nl) * K;

  v8f c0 = {}, c1 = {}, c2 = {}, c3 = {};
  for (int k0 = 0; k0 < K; k0 += 32) {
    v16bf af = ldA16(arow + k0 + khalf * 16, arow + k0 + khalf * 16 + 8);
    v16bf b0 = *(const v16bf*)(w0 + k0 + khalf * 16);
    v16bf b1 = *(const v16bf*)(w1 + k0 + khalf * 16);
    v16bf b2 = *(const v16bf*)(w2 + k0 + khalf * 16);
    v16bf b3 = *(const v16bf*)(w3 + k0 + khalf * 16);
    c0 = WMMA_BF16(af, b0, c0);
    c1 = WMMA_BF16(af, b1, c1);
    c2 = WMMA_BF16(af, b2, c2);
    c3 = WMMA_BF16(af, b3, c3);
  }
#pragma unroll
  for (int r = 0; r < 8; ++r) {
    float* crow = C + (size_t)(m0 + r + 8 * khalf) * N + n0 + nl;
    crow[0]  = c0[r];
    crow[16] = c1[r];
    crow[32] = c2[r];
    crow[48] = c3[r];
  }
}

// -------------------------------------------------------- 3D rotary embedding
// x f32 [B*L, 768] (heads interleaved) -> bf16 [B, H, L, 96]
__global__ void rope_kernel(const float* __restrict__ xf, const float* __restrict__ coords,
                            __bf16* __restrict__ xh, int L) {
  int i = blockIdx.x * blockDim.x + threadIdx.x;   // (b, l, h, j), j in [0,48)
  int j = i % 48; int t = i / 48;
  int h = t % 8;  t /= 8;
  int l = t % L;  int b = t / L;
  if (b >= 2) return;
  int axis = j >> 4, jj = j & 15;
  const float* row = xf + ((size_t)(b * L + l)) * 768 + h * 96 + axis * 32;
  float x1 = row[jj];
  float x2 = row[jj + 16];
  float c  = coords[((size_t)(b * L + l)) * 3 + axis];
  // inv_freq = 10000^(-jj/16) = exp(-jj * ln(10000)/16)
  float inv = __expf(-(float)jj * 0.5756462732485114f);
  float ang = c * inv;
  float sn = __sinf(ang), cs = __cosf(ang);
  __bf16* orow = xh + (((size_t)(b * 8 + h)) * L + l) * 96 + axis * 32;
  orow[jj]      = (__bf16)(x1 * cs - x2 * sn);
  orow[jj + 16] = (__bf16)(x1 * sn + x2 * cs);
}

// --------------------------------------- V: f32 [B*L,768] -> bf16 [B,H,L,96]
__global__ void split_v_kernel(const float* __restrict__ vf, __bf16* __restrict__ vh, int L) {
  int i = blockIdx.x * blockDim.x + threadIdx.x;
  if (i >= 2 * L * 768) return;
  int d = i % 96; int r = i / 96;
  int l = r % L;  int bh = r / L;
  int h = bh & 7; int b = bh >> 3;
  vh[i] = (__bf16)vf[((size_t)(b * L + l)) * 768 + h * 96 + d];
}

// ---------------------------------------------------------- flash attention
// grid = (L/64, B*H); 4 waves, each wave owns 16 query rows; 32 keys / step.
// K tile staged via GLOBAL_LOAD_ASYNC_TO_LDS_B128 (ASYNCcnt), V transposed
// manually into LDS for contiguous P*V B-fragments.
__global__ __launch_bounds__(128) void attn_kernel(
    const __bf16* __restrict__ qh, const __bf16* __restrict__ kh,
    const __bf16* __restrict__ vh, __bf16* __restrict__ ao, int L) {
  __shared__ __bf16 Kt[32 * 96];       // [key][d]  : row-major, contiguous frags
  __shared__ __bf16 Vt[96 * 32];       // [d][key]  : transposed for P*V frags
  __shared__ __bf16 Pst[4][16 * 32];   // per-wave P staging (C-layout -> A-frag)

  const int lane  = threadIdx.x & 31;
  const int wave  = threadIdx.x >> 5;
  const int nl    = lane & 15;
  const int khalf = lane >> 4;
  const int bh = blockIdx.y;
  const int b  = bh >> 3, h = bh & 7;
  const int qbase = blockIdx.x * 64 + wave * 16;
  const float scale = 0.10206207261596575f;  // 1/sqrt(96)

  // Q fragments for this wave's 16 rows (d = 96 -> 3 A-fragments), kept in VGPRs.
  const __bf16* qrow = qh + ((size_t)bh * L + qbase + nl) * 96;
  v16bf qf0 = ldA16(qrow + 0  + khalf * 16, qrow + 8  + khalf * 16);
  v16bf qf1 = ldA16(qrow + 32 + khalf * 16, qrow + 40 + khalf * 16);
  v16bf qf2 = ldA16(qrow + 64 + khalf * 16, qrow + 72 + khalf * 16);

  v8f o0 = {}, o1 = {}, o2 = {}, o3 = {}, o4 = {}, o5 = {};
  float mrow[8], lrow[8];
#pragma unroll
  for (int r = 0; r < 8; ++r) { mrow[r] = -1e30f; lrow[r] = 0.f; }

  const __bf16* kbase = kh + (size_t)bh * L * 96;
  const __bf16* vbase = vh + (size_t)bh * L * 96;
  __bf16* pw = &Pst[wave][0];

  for (int j0 = 0; j0 < L; j0 += 32) {
    __syncthreads();   // previous tile fully consumed
    // Stage K tile [32x96] via async global->LDS (no VGPR round-trip);
    // transpose V tile [32x96] -> Vt [96x32] through VGPRs.
#pragma unroll
    for (int i = 0; i < 3; ++i) {
      int chunk = threadIdx.x + 128 * i;        // 384 chunks of 8 bf16
      int row = chunk / 12, c8 = (chunk % 12) * 8;
      unsigned long long ga =
          (unsigned long long)(kbase + (size_t)(j0 + row) * 96 + c8);
      unsigned la = (unsigned)(unsigned long long)&Kt[row * 96 + c8];  // LDS byte addr
      asm volatile("global_load_async_to_lds_b128 %0, %1, off"
                   :: "v"(la), "v"(ga) : "memory");
      v8bf vv = *(const v8bf*)(vbase + (size_t)(j0 + row) * 96 + c8);
#pragma unroll
      for (int e = 0; e < 8; ++e) Vt[(c8 + e) * 32 + row] = vv[e];
    }
    if (j0 + 32 < L) {
      __builtin_prefetch(kbase + (size_t)(j0 + 32) * 96 + threadIdx.x * 8, 0, 0);
      __builtin_prefetch(vbase + (size_t)(j0 + 32) * 96 + threadIdx.x * 8, 0, 0);
    }
    asm volatile("s_wait_asynccnt 0x0" ::: "memory");
    __syncthreads();

    // S = Q * K^T for two 16-key sub-tiles (3 WMMAs each over d=96).
    v8f s0 = {}, s1 = {};
    {
      v16bf bk;
      bk = *(const v16bf*)&Kt[nl * 96        + 0  + khalf * 16]; s0 = WMMA_BF16(qf0, bk, s0);
      bk = *(const v16bf*)&Kt[nl * 96        + 32 + khalf * 16]; s0 = WMMA_BF16(qf1, bk, s0);
      bk = *(const v16bf*)&Kt[nl * 96        + 64 + khalf * 16]; s0 = WMMA_BF16(qf2, bk, s0);
      bk = *(const v16bf*)&Kt[(16 + nl) * 96 + 0  + khalf * 16]; s1 = WMMA_BF16(qf0, bk, s1);
      bk = *(const v16bf*)&Kt[(16 + nl) * 96 + 32 + khalf * 16]; s1 = WMMA_BF16(qf1, bk, s1);
      bk = *(const v16bf*)&Kt[(16 + nl) * 96 + 64 + khalf * 16]; s1 = WMMA_BF16(qf2, bk, s1);
    }
    s0 = s0 * scale;
    s1 = s1 * scale;

    // Online softmax. C-layout: register r = query row (r + 8*khalf),
    // lane = key column -> row stats reduce across 16-lane halves.
#pragma unroll
    for (int r = 0; r < 8; ++r) {
      float t = fmaxf(s0[r], s1[r]);
      t = fmaxf(t, __shfl_xor(t, 1, 16));
      t = fmaxf(t, __shfl_xor(t, 2, 16));
      t = fmaxf(t, __shfl_xor(t, 4, 16));
      t = fmaxf(t, __shfl_xor(t, 8, 16));
      float mn = fmaxf(mrow[r], t);
      float alpha = __expf(mrow[r] - mn);
      mrow[r] = mn;
      float p0 = __expf(s0[r] - mn);
      float p1 = __expf(s1[r] - mn);
      float rs = p0 + p1;
      rs += __shfl_xor(rs, 1, 16);
      rs += __shfl_xor(rs, 2, 16);
      rs += __shfl_xor(rs, 4, 16);
      rs += __shfl_xor(rs, 8, 16);
      lrow[r] = lrow[r] * alpha + rs;
      o0[r] *= alpha; o1[r] *= alpha; o2[r] *= alpha;
      o3[r] *= alpha; o4[r] *= alpha; o5[r] *= alpha;
      int prow = r + 8 * khalf;
      pw[prow * 32 + nl]      = (__bf16)p0;
      pw[prow * 32 + 16 + nl] = (__bf16)p1;
    }

    // P (16x32) A-fragment from LDS; O += P * V (6 d-tiles of 16).
    v16bf pf = ldA16(pw + nl * 32 + khalf * 16, pw + nl * 32 + 8 + khalf * 16);
    v16bf bv;
    bv = *(const v16bf*)&Vt[(0 * 16 + nl) * 32 + khalf * 16]; o0 = WMMA_BF16(pf, bv, o0);
    bv = *(const v16bf*)&Vt[(1 * 16 + nl) * 32 + khalf * 16]; o1 = WMMA_BF16(pf, bv, o1);
    bv = *(const v16bf*)&Vt[(2 * 16 + nl) * 32 + khalf * 16]; o2 = WMMA_BF16(pf, bv, o2);
    bv = *(const v16bf*)&Vt[(3 * 16 + nl) * 32 + khalf * 16]; o3 = WMMA_BF16(pf, bv, o3);
    bv = *(const v16bf*)&Vt[(4 * 16 + nl) * 32 + khalf * 16]; o4 = WMMA_BF16(pf, bv, o4);
    bv = *(const v16bf*)&Vt[(5 * 16 + nl) * 32 + khalf * 16]; o5 = WMMA_BF16(pf, bv, o5);
  }

  // Normalize and write bf16 in [B*L, 768] (heads re-interleaved) for Wo GEMM.
#pragma unroll
  for (int r = 0; r < 8; ++r) {
    float inv = 1.f / lrow[r];
    int row = qbase + r + 8 * khalf;
    __bf16* orow = ao + ((size_t)(b * L + row)) * 768 + h * 96;
    orow[0 * 16 + nl] = (__bf16)(o0[r] * inv);
    orow[1 * 16 + nl] = (__bf16)(o1[r] * inv);
    orow[2 * 16 + nl] = (__bf16)(o2[r] * inv);
    orow[3 * 16 + nl] = (__bf16)(o3[r] * inv);
    orow[4 * 16 + nl] = (__bf16)(o4[r] * inv);
    orow[5 * 16 + nl] = (__bf16)(o5[r] * inv);
  }
}

extern "C" void kernel_launch(void* const* d_in, const int* in_sizes, int n_in,
                              void* d_out, int out_size, void* d_ws, size_t ws_size,
                              hipStream_t stream) {
  (void)in_sizes; (void)n_in; (void)out_size; (void)ws_size;
  const int B = 2, L = 2048, DIM = 768, H = 8;
  const float* Q_in     = (const float*)d_in[0];
  const float* K_in     = (const float*)d_in[1];
  const float* V_in     = (const float*)d_in[2];
  const float* coords_q = (const float*)d_in[3];
  const float* coords_k = (const float*)d_in[4];
  const float* Wq       = (const float*)d_in[5];
  const float* Wk       = (const float*)d_in[6];
  const float* Wv       = (const float*)d_in[7];
  const float* Wo       = (const float*)d_in[8];

  const size_t nX = (size_t)B * L * DIM;   // 3,145,728
  const size_t nW = (size_t)DIM * DIM;     //   589,824
  const size_t nH = (size_t)B * H * L * 96;

  char* ws = (char*)d_ws;
  __bf16* Qb  = (__bf16*)ws; ws += nX * sizeof(__bf16);
  __bf16* Kb  = (__bf16*)ws; ws += nX * sizeof(__bf16);
  __bf16* Vb  = (__bf16*)ws; ws += nX * sizeof(__bf16);
  __bf16* Wqb = (__bf16*)ws; ws += nW * sizeof(__bf16);
  __bf16* Wkb = (__bf16*)ws; ws += nW * sizeof(__bf16);
  __bf16* Wvb = (__bf16*)ws; ws += nW * sizeof(__bf16);
  __bf16* Wob = (__bf16*)ws; ws += nW * sizeof(__bf16);
  float*  qf  = (float*)ws;  ws += nX * sizeof(float);
  float*  kf  = (float*)ws;  ws += nX * sizeof(float);
  float*  vf  = (float*)ws;  ws += nX * sizeof(float);
  __bf16* qhb = (__bf16*)ws; ws += nH * sizeof(__bf16);
  __bf16* khb = (__bf16*)ws; ws += nH * sizeof(__bf16);
  __bf16* vhb = (__bf16*)ws; ws += nH * sizeof(__bf16);
  __bf16* aob = (__bf16*)ws; ws += nX * sizeof(__bf16);

  // 1) convert inputs + weights to bf16
  cvt_bf16_kernel<<<(nX + 255) / 256, 256, 0, stream>>>(Q_in, Qb, (int)nX);
  cvt_bf16_kernel<<<(nX + 255) / 256, 256, 0, stream>>>(K_in, Kb, (int)nX);
  cvt_bf16_kernel<<<(nX + 255) / 256, 256, 0, stream>>>(V_in, Vb, (int)nX);
  cvt_bf16_kernel<<<(nW + 255) / 256, 256, 0, stream>>>(Wq, Wqb, (int)nW);
  cvt_bf16_kernel<<<(nW + 255) / 256, 256, 0, stream>>>(Wk, Wkb, (int)nW);
  cvt_bf16_kernel<<<(nW + 255) / 256, 256, 0, stream>>>(Wv, Wvb, (int)nW);
  cvt_bf16_kernel<<<(nW + 255) / 256, 256, 0, stream>>>(Wo, Wob, (int)nW);

  // 2) projections: [4096,768] = X * W^T (WMMA bf16 -> f32), 128x64 block tiles
  dim3 ggrid(DIM / 64, (B * L) / 128);
  gemm_bf16_f32_kernel<<<ggrid, 256, 0, stream>>>(Qb, Wqb, qf, B * L, DIM, DIM);
  gemm_bf16_f32_kernel<<<ggrid, 256, 0, stream>>>(Kb, Wkb, kf, B * L, DIM, DIM);
  gemm_bf16_f32_kernel<<<ggrid, 256, 0, stream>>>(Vb, Wvb, vf, B * L, DIM, DIM);

  // 3) RoPE on q/k (f32 math -> bf16 [B,H,L,96]), V head-split
  int ropeN = B * L * H * 48;
  rope_kernel<<<(ropeN + 255) / 256, 256, 0, stream>>>(qf, coords_q, qhb, L);
  rope_kernel<<<(ropeN + 255) / 256, 256, 0, stream>>>(kf, coords_k, khb, L);
  split_v_kernel<<<((int)nX + 255) / 256, 256, 0, stream>>>(vf, vhb, L);

  // 4) flash attention per (b,h): 64 query rows / block
  dim3 agrid(L / 64, B * H);
  attn_kernel<<<agrid, 128, 0, stream>>>(qhb, khb, vhb, aob, L);

  // 5) output projection straight into d_out (f32)
  gemm_bf16_f32_kernel<<<ggrid, 256, 0, stream>>>(aob, Wob, (float*)d_out, B * L, DIM, DIM);
}
